// DeformableConv2d_4964982194859
// MI455X (gfx1250) — matrix operand order repaired and checked by reading, not compiled, not run on hardware
//
#include <hip/hip_runtime.h>
#include <hip/hip_bf16.h>

// ---------------------------------------------------------------------------
// Deformable Conv2d for gfx1250 (MI455X), all-WMMA bf16 path.
// B=4, Cin=Cout=64, H=W=128, K=3, PAD=1, STRIDE=1.
// Both GEMMs (offset/mask conv M=27 and main conv M=64, K=576, N=65536)
// run on v_wmma_f32_16x16x32_bf16 with fp32 accumulation.
// ---------------------------------------------------------------------------

typedef __attribute__((ext_vector_type(16))) __bf16 v16bf;
typedef __attribute__((ext_vector_type(8)))  float  v8f;

#define BB    4
#define CIN   64
#define COUT  64
#define HH    128
#define WW    128
#define HWSZ  (HH * WW)          // 16384
#define K2V   9
#define KDIM  (CIN * K2V)        // 576
#define TSTEPS (KDIM / 32)       // 18 k-steps of 32
#define NTILE 64                 // output positions per block
#define OMC   27                 // 3*K2 offset/mask channels

// A-fragment tables in workspace (bf16):
//   main:  4 m-groups * 18 t * 32 lanes * 16 elems = 36864
//   om  :  2 m-groups * 18 t * 32 lanes * 16 elems = 18432 (rows >=27 zero)
#define WFRAG_MAIN_ELEMS (4 * TSTEPS * 32 * 16)
#define WFRAG_OM_ELEMS   (2 * TSTEPS * 32 * 16)

// ---------------------------------------------------------------------------
// Kernel 0: shuffle fp32 weights into bf16 WMMA A-fragments.
// A 16x32 bf16 layout: lane half=lane>>4, m=lane&15;
//   K = t*32 + half*16 + j
// ---------------------------------------------------------------------------
__global__ void dcn_prep_wfrag(const float* __restrict__ w,      // [64][576]
                               const float* __restrict__ omw,    // [27][576]
                               __bf16* __restrict__ wfrag,
                               __bf16* __restrict__ omwfrag) {
    int e = blockIdx.x * blockDim.x + threadIdx.x;
    if (e >= WFRAG_MAIN_ELEMS + WFRAG_OM_ELEMS) return;

    int isOm = (e >= WFRAG_MAIN_ELEMS);
    int ee   = isOm ? (e - WFRAG_MAIN_ELEMS) : e;

    int j    = ee & 15;
    int lane = (ee >> 4) & 31;
    int t    = (ee >> 9) % TSTEPS;
    int g    = ee / (TSTEPS * 512);
    int m    = g * 16 + (lane & 15);
    int half = lane >> 4;
    int kidx = t * 32 + half * 16 + j;
    if (isOm) {
        float v = (m < OMC) ? omw[m * KDIM + kidx] : 0.0f;
        omwfrag[ee] = (__bf16)v;
    } else {
        wfrag[ee] = (__bf16)w[m * KDIM + kidx];
    }
}

// ---------------------------------------------------------------------------
// Kernel 1: offset/mask conv as im2col GEMM on WMMA.
// Block: 256 threads = 8 waves -> 32 channels (27 used) x 64 positions.
// Waves: mg = wv&1 (16 ch), ng = wv>>1 (16 positions), 18 WMMAs each.
// ---------------------------------------------------------------------------
__global__ __launch_bounds__(256)
void dcn_om_wmma(const float* __restrict__ x,
                 const float* __restrict__ omb,
                 const __bf16* __restrict__ omwfrag,
                 float* __restrict__ om) {
    __shared__ __attribute__((aligned(32))) __bf16 S[TSTEPS * NTILE * 32]; // 72 KB

    const int tid     = threadIdx.x;
    const int posBase = blockIdx.x * NTILE;
    const int b       = posBase >> 14;
    const int hw      = posBase & (HWSZ - 1);
    const int h       = hw >> 7;
    const int wbase   = hw & (WW - 1);

    // ---- Phase 0: zero-padded im2col gather -> bf16 B-fragment LDS --------
    // S[r][n] = x[b, c, h-1+ky, wbase+n-1+kx], r = c*9 + ky*3 + kx.
    for (int e = tid; e < KDIM * NTILE; e += 256) {
        int n = e & (NTILE - 1);
        int r = e >> 6;
        int c = r / K2V;
        int k = r - c * K2V;
        int y  = h - 1 + (k / 3);
        int xx = wbase + n - 1 + (k % 3);
        float v = 0.0f;
        if ((y >= 0) & (y < HH) & (xx >= 0) & (xx < WW))
            v = x[(size_t)(b * CIN + c) * HWSZ + y * WW + xx];
        S[(r >> 5) * (NTILE * 32) + n * 32 + (r & 31)] = (__bf16)v;
    }
    __syncthreads();

    // ---- Phase 1: WMMA GEMM [32 x 64] = [32 x 576] x [576 x 64] -----------
    const int wv   = tid >> 5;
    const int lane = tid & 31;
    const int mg   = wv & 1;            // channel group: mg*16..+16
    const int ng   = wv >> 1;           // position group: ng*16..+16
    const int half = lane >> 4;
    const int col  = lane & 15;

    v8f acc = {};
    const v16bf* Wv = (const v16bf*)omwfrag;
    for (int t = 0; t < TSTEPS; ++t) {
        v16bf a = Wv[(mg * TSTEPS + t) * 32 + lane];
        const v16bf* bf =
            (const v16bf*)&S[t * (NTILE * 32) + (ng * 16 + col) * 32 + half * 16];
        acc = __builtin_amdgcn_wmma_f32_16x16x32_bf16(
            false, a, false, *bf, (short)0, acc, false, false);
    }

    // ---- Epilogue: om[b][o][h][w] + om_bias, skip padded rows o>=27 -------
#pragma unroll
    for (int r8 = 0; r8 < 8; ++r8) {
        int o = mg * 16 + r8 + half * 8;
        if (o < OMC) {
            int w = wbase + ng * 16 + col;
            om[(((size_t)(b * OMC + o) * HH + h) * WW) + w] = acc[r8] + omb[o];
        }
    }
}

// ---------------------------------------------------------------------------
// Kernel 2: fused bilinear sampling + bf16 WMMA GEMM (main conv).
// Block: 256 threads = 8 waves -> 64 channels x 64 positions.
// ---------------------------------------------------------------------------
__global__ __launch_bounds__(256)
void dcn_main(const float* __restrict__ x,
              const float* __restrict__ bias,
              const float* __restrict__ om,
              const __bf16* __restrict__ wfrag,
              float* __restrict__ out) {
    __shared__ int   dsc_i[K2V * NTILE * 4];                // 4 corner indices
    __shared__ float dsc_w[K2V * NTILE * 4];                // corner weights * mask
    __shared__ __attribute__((aligned(32))) __bf16 S[TSTEPS * NTILE * 32]; // 72 KB

    const int tid     = threadIdx.x;
    const int posBase = blockIdx.x * NTILE;    // 64-aligned -> single (b,h) row
    const int b       = posBase >> 14;
    const int hw      = posBase & (HWSZ - 1);
    const int h       = hw >> 7;
    const int wbase   = hw & (WW - 1);

    // ---- Phase 0: bilinear descriptors for 9 taps x 64 positions ----------
    for (int p = tid; p < K2V * NTILE; p += 256) {
        int k = p >> 6;                 // tap 0..8
        int n = p & (NTILE - 1);
        int w = wbase + n;
        const float* omp = om + (((size_t)(b * OMC + k) * HH + h) * WW + w);
        float dy = omp[0];
        float dx = omp[(size_t)K2V * HWSZ];
        float mm = omp[(size_t)(2 * K2V) * HWSZ];
        float mask = 1.0f / (1.0f + __expf(-mm));
        float py = dy + (float)(k / 3) + (float)(h - 1);
        float px = dx + (float)(k % 3) + (float)(w - 1);
        float y0f = floorf(py), x0f = floorf(px);
        float wy = py - y0f,    wx = px - x0f;
        int y0 = (int)y0f, x0 = (int)x0f;
        int y1 = y0 + 1,   x1 = x0 + 1;
        bool vy0 = (y0 >= 0) & (y0 < HH), vy1 = (y1 >= 0) & (y1 < HH);
        bool vx0 = (x0 >= 0) & (x0 < WW), vx1 = (x1 >= 0) & (x1 < WW);
        float w00 = (1.f - wy) * (1.f - wx) * mask; if (!(vy0 && vx0)) w00 = 0.f;
        float w01 = (1.f - wy) * wx         * mask; if (!(vy0 && vx1)) w01 = 0.f;
        float w10 = wy * (1.f - wx)         * mask; if (!(vy1 && vx0)) w10 = 0.f;
        float w11 = wy * wx                 * mask; if (!(vy1 && vx1)) w11 = 0.f;
        int y0c = min(max(y0, 0), HH - 1), y1c = min(max(y1, 0), HH - 1);
        int x0c = min(max(x0, 0), WW - 1), x1c = min(max(x1, 0), WW - 1);
        dsc_i[p * 4 + 0] = y0c * WW + x0c;
        dsc_i[p * 4 + 1] = y0c * WW + x1c;
        dsc_i[p * 4 + 2] = y1c * WW + x0c;
        dsc_i[p * 4 + 3] = y1c * WW + x1c;
        dsc_w[p * 4 + 0] = w00;
        dsc_w[p * 4 + 1] = w01;
        dsc_w[p * 4 + 2] = w10;
        dsc_w[p * 4 + 3] = w11;
    }
    __syncthreads();

    // ---- Phase 1: sample S[r=c*9+k][n] -> LDS in B-fragment order ---------
    for (int e = tid; e < KDIM * NTILE; e += 256) {
        int n = e & (NTILE - 1);
        int r = e >> 6;                 // c*9 + k
        int c = r / K2V;
        int k = r - c * K2V;
        int p = k * NTILE + n;
        int i00 = dsc_i[p * 4 + 0], i01 = dsc_i[p * 4 + 1];
        int i10 = dsc_i[p * 4 + 2], i11 = dsc_i[p * 4 + 3];
        float w00 = dsc_w[p * 4 + 0], w01 = dsc_w[p * 4 + 1];
        float w10 = dsc_w[p * 4 + 2], w11 = dsc_w[p * 4 + 3];
        const float* xp = x + (size_t)(b * CIN + c) * HWSZ;
        float v = w00 * xp[i00] + w01 * xp[i01] + w10 * xp[i10] + w11 * xp[i11];
        S[(r >> 5) * (NTILE * 32) + n * 32 + (r & 31)] = (__bf16)v;
    }
    __syncthreads();

    // ---- Phase 2: D[64x64] = W[64x576] x S[576x64] via bf16 WMMA ----------
    const int wv   = tid >> 5;
    const int lane = tid & 31;
    const int mg   = wv & 3;            // channel group: mg*16..+16
    const int ng   = wv >> 2;           // position group: ng*32..+32
    const int half = lane >> 4;
    const int col  = lane & 15;

    v8f acc0 = {};
    v8f acc1 = {};
    const v16bf* Wv = (const v16bf*)wfrag;
    for (int t = 0; t < TSTEPS; ++t) {
        v16bf a = Wv[(mg * TSTEPS + t) * 32 + lane];
        const v16bf* b0 =
            (const v16bf*)&S[t * (NTILE * 32) + (ng * 32 + col) * 32 + half * 16];
        const v16bf* b1 =
            (const v16bf*)&S[t * (NTILE * 32) + (ng * 32 + 16 + col) * 32 + half * 16];
        acc0 = __builtin_amdgcn_wmma_f32_16x16x32_bf16(
            false, a, false, *b0, (short)0, acc0, false, false);
        acc1 = __builtin_amdgcn_wmma_f32_16x16x32_bf16(
            false, a, false, *b1, (short)0, acc1, false, false);
    }

    // ---- Epilogue: C/D layout -> out[b][ch][h][w] + bias ------------------
#pragma unroll
    for (int r8 = 0; r8 < 8; ++r8) {
        int ch = mg * 16 + r8 + half * 8;
        float bv = bias[ch];
        int n0 = ng * 32 + col;
        int n1 = n0 + 16;
        size_t base = (size_t)(b * COUT + ch) * HWSZ;
        out[base + ((posBase + n0) & (HWSZ - 1))] = acc0[r8] + bv;
        out[base + ((posBase + n1) & (HWSZ - 1))] = acc1[r8] + bv;
    }
}

// ---------------------------------------------------------------------------
// Launch. d_in order: x, weight, bias, om_weight, om_bias.
// d_ws layout: [0, 73728) main A-fragments (bf16)
//              [73728, 110592) om A-fragments (bf16)
//              [110592, +7.08MB) fp32 om buffer
// ---------------------------------------------------------------------------
extern "C" void kernel_launch(void* const* d_in, const int* in_sizes, int n_in,
                              void* d_out, int out_size, void* d_ws, size_t ws_size,
                              hipStream_t stream) {
    const float* x       = (const float*)d_in[0];
    const float* weight  = (const float*)d_in[1];
    const float* bias    = (const float*)d_in[2];
    const float* om_w    = (const float*)d_in[3];
    const float* om_b    = (const float*)d_in[4];
    float* out           = (float*)d_out;

    __bf16* wfrag   = (__bf16*)d_ws;
    __bf16* omwfrag = (__bf16*)((char*)d_ws + WFRAG_MAIN_ELEMS * 2);
    float*  om      = (float*)((char*)d_ws +
                               (WFRAG_MAIN_ELEMS + WFRAG_OM_ELEMS) * 2);

    // Kernel 0: weight shuffle (main + om) to bf16 A-fragments.
    {
        int total = WFRAG_MAIN_ELEMS + WFRAG_OM_ELEMS;    // 55296
        dcn_prep_wfrag<<<(total + 255) / 256, 256, 0, stream>>>(
            weight, om_w, wfrag, omwfrag);
    }
    // Kernel 1: offset/mask conv as WMMA im2col GEMM (1024 blocks).
    {
        int blocks = (BB * HWSZ) / NTILE;   // 1024
        dcn_om_wmma<<<blocks, 256, 0, stream>>>(x, om_b, omwfrag, om);
    }
    // Kernel 2: fused sampling + WMMA GEMM (1024 blocks).
    {
        int blocks = (BB * HWSZ) / NTILE;   // 1024
        dcn_main<<<blocks, 256, 0, stream>>>(x, bias, om, wfrag, out);
    }
}
